// GraphClsGAT_52621939310630
// MI455X (gfx1250) — matrix operand order repaired and checked by reading, not compiled, not run on hardware
//
#include <hip/hip_runtime.h>
#include <hip/hip_bf16.h>

// ---------------------------------------------------------------------------
// GAT (2 layers, 4 heads x 64) + linear classifier for MI455X (gfx1250).
// GEMMs: bf16 WMMA (v_wmma_f32_16x16x32_bf16, f32 accumulate) fed DIRECTLY
// from global memory with b128 loads (X row-major, W pre-transposed), no LDS
// round trip, no barriers. Each wave: 2 M-subtiles x 4 N-subtiles = 8 WMMAs
// per 32-deep k-step. Fused el/er head projections via wave32 shfl_xor.
// Edge softmax/aggregation: f32 global atomics.
// ---------------------------------------------------------------------------

typedef __bf16 bf16_t;
typedef __attribute__((ext_vector_type(8)))  __bf16 v8bf;
typedef __attribute__((ext_vector_type(16))) __bf16 v16bf;
typedef __attribute__((ext_vector_type(8)))  float  v8f;

#define N_NODES   30000
#define N_EDGES   480000
#define FEATS     256        // both layers: 256 in -> 256 out (4 heads x 64)
#define HEADS     4
#define HDIM      64
#define NCLASS    10
#define NEG_SLOPE 0.2f

// ---------------------------------------------------------------------------
// elementwise helpers
// ---------------------------------------------------------------------------
__global__ __launch_bounds__(256)
void k_f32_to_bf16(const float* __restrict__ in, bf16_t* __restrict__ out, int n) {
  int i = blockIdx.x * blockDim.x + threadIdx.x;
  if (i < n) out[i] = (bf16_t)in[i];
}

// Wt[n][k] = (bf16) W[k][n]  — makes B fragments K-contiguous in memory.
__global__ __launch_bounds__(256)
void k_transpose_bf16(const float* __restrict__ W, bf16_t* __restrict__ Wt) {
  int t = blockIdx.x * blockDim.x + threadIdx.x;   // 0..65535
  int n = t >> 8, k = t & (FEATS - 1);
  Wt[t] = (bf16_t)W[k * FEATS + n];
}

__global__ __launch_bounds__(256)
void k_fill(float* __restrict__ p, float v, int n) {
  int i = blockIdx.x * blockDim.x + threadIdx.x;
  if (i < n) p[i] = v;
}

// ---------------------------------------------------------------------------
// Fused GEMM + head projection:
//   Z[M x 256] = X[M x 256](bf16) @ W[256 x 256](bf16)   (f32 accumulate)
//   el[n,h] = <Z[n,h,:], al[h,:]> ,  er likewise.
// Block = 128 threads (4 waves); block tile = 128 rows x 64 cols. Since
// HDIM == 64, blockIdx.x is exactly the head index. Wave tile = 32 x 64:
// 2 M-subtiles x 4 N-subtiles = 8 accumulators, 8 WMMAs per k-step.
// Fragments per ISA 16-bit layout tables:
//   A 16x32: lane(0-15)=rows, K base 0 / lane(16-31) K base 8; pairs packed.
//   B 32x16: lane%16 = N, K base 0 (lanes<16) / 8 (lanes>=16); same K order as A.
// Both map to two contiguous 16B global loads per fragment.
// ---------------------------------------------------------------------------
__global__ __launch_bounds__(128)
void k_gemm_gat(const bf16_t* __restrict__ X, const bf16_t* __restrict__ Wt,
                float* __restrict__ Z, const float* __restrict__ al,
                const float* __restrict__ ar, float* __restrict__ el,
                float* __restrict__ er, int M) {
  const int h    = blockIdx.x;            // head index == 64-col slice
  const int n0   = h * HDIM;
  const int m0   = blockIdx.y * 128;
  const int lane = threadIdx.x & 31;
  const int wv   = threadIdx.x >> 5;
  const int laneM = lane & 15;
  const int kba   = (lane < 16) ? 0 : 8;   // A-fragment K base
  const int kbb   = (lane < 16) ? 0 : 8;   // B-fragment K base

  // A row pointers for the two M-subtiles (clamped; stores guarded later).
  const bf16_t* aptr[2];
#pragma unroll
  for (int u = 0; u < 2; ++u) {
    int r = m0 + 32 * wv + 16 * u + laneM;
    if (r >= M) r = M - 1;
    aptr[u] = X + (size_t)r * FEATS + kba;
  }
  // B row pointers for the four N-subtiles (shared by all waves: L0-resident).
  const bf16_t* bptr[4];
#pragma unroll
  for (int t = 0; t < 4; ++t)
    bptr[t] = Wt + (size_t)(n0 + 16 * t + laneM) * FEATS + kbb;

  v8f acc[2][4] = {{{}, {}, {}, {}}, {{}, {}, {}, {}}};

#pragma unroll
  for (int k0 = 0; k0 < FEATS; k0 += 32) {
    v16bf av[2], bv[4];
#pragma unroll
    for (int u = 0; u < 2; ++u) {
      v8bf a0 = *(const v8bf*)(aptr[u] + k0);
      v8bf a1 = *(const v8bf*)(aptr[u] + k0 + 16);
#pragma unroll
      for (int i = 0; i < 8; ++i) { av[u][i] = a0[i]; av[u][8 + i] = a1[i]; }
    }
#pragma unroll
    for (int t = 0; t < 4; ++t) {
      v8bf b0 = *(const v8bf*)(bptr[t] + k0);
      v8bf b1 = *(const v8bf*)(bptr[t] + k0 + 16);
#pragma unroll
      for (int i = 0; i < 8; ++i) { bv[t][i] = b0[i]; bv[t][8 + i] = b1[i]; }
    }
#pragma unroll
    for (int u = 0; u < 2; ++u)
#pragma unroll
      for (int t = 0; t < 4; ++t)
        acc[u][t] = __builtin_amdgcn_wmma_f32_16x16x32_bf16(
            false, av[u], false, bv[t], (short)0, acc[u][t], false, false);
  }

  // ---- store Z (C layout: VGPR v -> M = v + 8*(lane>=16); lane%16 -> N) ----
  const int mhalf  = (lane < 16) ? 0 : 8;
  const int outCol = n0 + laneM;
  if (m0 + 128 <= M) {                    // fast path: full tile, no guards
#pragma unroll
    for (int u = 0; u < 2; ++u)
#pragma unroll
      for (int v = 0; v < 8; ++v) {
        float* zp =
            Z + (m0 + 32 * wv + 16 * u + mhalf + v) * FEATS + outCol;
#pragma unroll
        for (int t = 0; t < 4; ++t) zp[16 * t] = acc[u][t][v];
      }
  } else {
#pragma unroll
    for (int u = 0; u < 2; ++u)
#pragma unroll
      for (int v = 0; v < 8; ++v) {
        int row = m0 + 32 * wv + 16 * u + mhalf + v;
        if (row < M) {
          float* zp = Z + row * FEATS + outCol;
#pragma unroll
          for (int t = 0; t < 4; ++t) zp[16 * t] = acc[u][t][v];
        }
      }
  }

  // ---- fused head projection: el/er for this head ----
  float alv[4], arv[4];
#pragma unroll
  for (int t = 0; t < 4; ++t) {
    alv[t] = al[h * HDIM + 16 * t + laneM];
    arv[t] = ar[h * HDIM + 16 * t + laneM];
  }
#pragma unroll
  for (int u = 0; u < 2; ++u)
#pragma unroll
    for (int v = 0; v < 8; ++v) {
      float pl = 0.f, pr = 0.f;
#pragma unroll
      for (int t = 0; t < 4; ++t) {
        float z = acc[u][t][v];
        pl += z * alv[t];
        pr += z * arv[t];
      }
      // reduce over the 16 lanes of each half-wave (cols of this head)
#pragma unroll
      for (int off = 1; off < 16; off <<= 1) {
        pl += __shfl_xor(pl, off, 32);
        pr += __shfl_xor(pr, off, 32);
      }
      int row = m0 + 32 * wv + 16 * u + mhalf + v;
      if (laneM == 0 && row < M) {        // lanes 0 and 16 hold the sums
        el[row * HEADS + h] = pl;
        er[row * HEADS + h] = pr;
      }
    }
}

// ---------------------------------------------------------------------------
// float atomic max via signed/unsigned int ordering trick.
// ---------------------------------------------------------------------------
__device__ inline void atomicMaxFloat(float* addr, float value) {
  if (value >= 0.f)
    atomicMax((int*)addr, __float_as_int(value));
  else
    atomicMin((unsigned int*)addr, __float_as_uint(value));
}

// e = leaky_relu(el[src]+er[dst]); stash e; segment-max into m[dst].
__global__ __launch_bounds__(256)
void k_edge_max(const int* __restrict__ src, const int* __restrict__ dst,
                const float* __restrict__ el, const float* __restrict__ er,
                float* __restrict__ ebuf, float* __restrict__ m) {
  int t = blockIdx.x * blockDim.x + threadIdx.x;
  if (t >= N_EDGES * HEADS) return;
  int e = t >> 2, h = t & 3;
  int s = src[e], d = dst[e];
  float v = el[s * HEADS + h] + er[d * HEADS + h];
  v = (v > 0.f) ? v : NEG_SLOPE * v;
  ebuf[t] = v;
  atomicMaxFloat(&m[d * HEADS + h], v);
}

// ex = exp(e - m[dst]); stash ex; segment-sum into denom[dst].
__global__ __launch_bounds__(256)
void k_edge_exp(const int* __restrict__ dst, float* __restrict__ ebuf,
                const float* __restrict__ m, float* __restrict__ denom) {
  int t = blockIdx.x * blockDim.x + threadIdx.x;
  if (t >= N_EDGES * HEADS) return;
  int e = t >> 2, h = t & 3;
  int d = dst[e];
  float ex = __expf(ebuf[t] - m[d * HEADS + h]);
  ebuf[t] = ex;
  atomicAdd(&denom[d * HEADS + h], ex);
}

// out[dst, c] += Z[src, c] * alpha ; one block per edge, 256 channels.
__global__ __launch_bounds__(256)
void k_edge_agg(const int* __restrict__ src, const int* __restrict__ dst,
                const float* __restrict__ ebuf, const float* __restrict__ denom,
                const float* __restrict__ Z, float* __restrict__ out) {
  int e = blockIdx.x;
  int c = threadIdx.x;
  int h = c >> 6;
  int s = src[e], d = dst[e];
  float alpha = ebuf[e * HEADS + h] / (denom[d * HEADS + h] + 1e-9f);
  atomicAdd(&out[d * FEATS + c], Z[s * FEATS + c] * alpha);
}

// h = elu(out + b), written as bf16 to feed the next WMMA GEMM.
__global__ __launch_bounds__(256)
void k_bias_elu_bf16(const float* __restrict__ acc, const float* __restrict__ b,
                     bf16_t* __restrict__ hout) {
  int t = blockIdx.x * blockDim.x + threadIdx.x;
  if (t >= N_NODES * FEATS) return;
  int c = t & (FEATS - 1);
  float v = acc[t] + b[c];
  v = (v > 0.f) ? v : expm1f(v);
  hout[t] = (bf16_t)v;
}

// logits = H(bf16) @ Wout + bout ; one wave32 per node, 8 nodes per block.
__global__ __launch_bounds__(256)
void k_classifier(const bf16_t* __restrict__ H, const float* __restrict__ Wout,
                  const float* __restrict__ bout, float* __restrict__ out) {
  int n = blockIdx.x * 8 + (threadIdx.x >> 5);
  int lane = threadIdx.x & 31;
  float p[NCLASS];
#pragma unroll
  for (int k = 0; k < NCLASS; ++k) p[k] = 0.f;
#pragma unroll
  for (int j = 0; j < FEATS / 32; ++j) {
    int c = lane + 32 * j;
    float x = (float)H[n * FEATS + c];
#pragma unroll
    for (int k = 0; k < NCLASS; ++k) p[k] += x * Wout[c * NCLASS + k];
  }
#pragma unroll
  for (int k = 0; k < NCLASS; ++k) {
    float v = p[k];
#pragma unroll
    for (int off = 16; off > 0; off >>= 1) v += __shfl_xor(v, off, 32);
    if (lane == 0) out[n * NCLASS + k] = v + bout[k];
  }
}

// ---------------------------------------------------------------------------
// Host orchestration.
// ---------------------------------------------------------------------------
namespace {

// workspace layout (bytes); all offsets 16B-aligned
constexpr size_t OFF_XBF  = 0;                                           // 30000*256*2
constexpr size_t OFF_WBF  = OFF_XBF  + (size_t)N_NODES * FEATS * 2;      // 256*256*2
constexpr size_t OFF_Z    = OFF_WBF  + (size_t)FEATS * FEATS * 2;        // 30000*256*4
constexpr size_t OFF_EL   = OFF_Z    + (size_t)N_NODES * FEATS * 4;      // 30000*4*4
constexpr size_t OFF_ER   = OFF_EL   + (size_t)N_NODES * HEADS * 4;
constexpr size_t OFF_M    = OFF_ER   + (size_t)N_NODES * HEADS * 4;
constexpr size_t OFF_DEN  = OFF_M    + (size_t)N_NODES * HEADS * 4;
constexpr size_t OFF_EBUF = OFF_DEN  + (size_t)N_NODES * HEADS * 4;      // 480000*4*4
constexpr size_t OFF_OUT  = OFF_EBUF + (size_t)N_EDGES * HEADS * 4;      // 30000*256*4

inline int cdiv(int a, int b) { return (a + b - 1) / b; }

void run_gat_layer(bf16_t* Xbf /*in: bf16 features, overwritten with layer out*/,
                   const float* W, const float* al, const float* ar,
                   const float* b, const int* src, const int* dst,
                   bf16_t* Wt, float* Z, float* el, float* er, float* m,
                   float* den, float* ebuf, float* accum, hipStream_t stream) {
  // W -> bf16, transposed to [n][k] so B fragments are K-contiguous
  k_transpose_bf16<<<cdiv(FEATS * FEATS, 256), 256, 0, stream>>>(W, Wt);
  // Z = Xbf @ W (WMMA, direct global feeds) with fused el/er projections
  dim3 ggrid(HEADS, cdiv(N_NODES, 128));
  k_gemm_gat<<<ggrid, 128, 0, stream>>>(Xbf, Wt, Z, al, ar, el, er, N_NODES);
  // init segment buffers
  k_fill<<<cdiv(N_NODES * HEADS, 256), 256, 0, stream>>>(m, -1e30f,
                                                         N_NODES * HEADS);
  k_fill<<<cdiv(N_NODES * HEADS, 256), 256, 0, stream>>>(den, 0.f,
                                                         N_NODES * HEADS);
  k_fill<<<cdiv(N_NODES * FEATS, 256), 256, 0, stream>>>(accum, 0.f,
                                                         N_NODES * FEATS);
  // edge softmax + aggregation
  k_edge_max<<<cdiv(N_EDGES * HEADS, 256), 256, 0, stream>>>(src, dst, el, er,
                                                             ebuf, m);
  k_edge_exp<<<cdiv(N_EDGES * HEADS, 256), 256, 0, stream>>>(dst, ebuf, m, den);
  k_edge_agg<<<N_EDGES, 256, 0, stream>>>(src, dst, ebuf, den, Z, accum);
  // bias + ELU -> bf16 (feeds next GEMM / classifier)
  k_bias_elu_bf16<<<cdiv(N_NODES * FEATS, 256), 256, 0, stream>>>(accum, b,
                                                                  Xbf);
}

}  // namespace

extern "C" void kernel_launch(void* const* d_in, const int* in_sizes, int n_in,
                              void* d_out, int out_size, void* d_ws,
                              size_t ws_size, hipStream_t stream) {
  const float* feat = (const float*)d_in[0];
  const int*   src  = (const int*)d_in[1];
  const int*   dst  = (const int*)d_in[2];
  const float* W1   = (const float*)d_in[3];
  const float* al1  = (const float*)d_in[4];
  const float* ar1  = (const float*)d_in[5];
  const float* b1   = (const float*)d_in[6];
  const float* W2   = (const float*)d_in[7];
  const float* al2  = (const float*)d_in[8];
  const float* ar2  = (const float*)d_in[9];
  const float* b2   = (const float*)d_in[10];
  const float* Wout = (const float*)d_in[11];
  const float* bout = (const float*)d_in[12];
  float* logits = (float*)d_out;

  char* ws = (char*)d_ws;
  bf16_t* Xbf  = (bf16_t*)(ws + OFF_XBF);
  bf16_t* Wt   = (bf16_t*)(ws + OFF_WBF);
  float*  Z    = (float*)(ws + OFF_Z);
  float*  el   = (float*)(ws + OFF_EL);
  float*  er   = (float*)(ws + OFF_ER);
  float*  m    = (float*)(ws + OFF_M);
  float*  den  = (float*)(ws + OFF_DEN);
  float*  ebuf = (float*)(ws + OFF_EBUF);
  float*  acc  = (float*)(ws + OFF_OUT);

  // feat -> bf16
  k_f32_to_bf16<<<cdiv(N_NODES * FEATS, 256), 256, 0, stream>>>(
      feat, Xbf, N_NODES * FEATS);

  // layer 1 (Xbf -> Xbf)
  run_gat_layer(Xbf, W1, al1, ar1, b1, src, dst, Wt, Z, el, er, m, den, ebuf,
                acc, stream);
  // layer 2 (Xbf -> Xbf)
  run_gat_layer(Xbf, W2, al2, ar2, b2, src, dst, Wt, Z, el, er, m, den, ebuf,
                acc, stream);
  // classifier
  k_classifier<<<N_NODES / 8, 256, 0, stream>>>(Xbf, Wout, bout, logits);
}
